// HybridQLSTM_65481071408028
// MI455X (gfx1250) — matrix operand order repaired and checked by reading, not compiled, not run on hardware
//
#include <hip/hip_runtime.h>
#include <hip/hip_bf16.h>
#include <math.h>

typedef __attribute__((ext_vector_type(16))) _Float16 v16h;
typedef __attribute__((ext_vector_type(8)))  float    v8f;

// ---------------------------------------------------------------------------
// Kernel 0: pack the x-part of Wf/Wi/Wu/Wo (each [4,1028], we use cols 0..1023)
// into the WMMA B-operand (32x16 f16, KxN) lane layout, one v16h per lane per
// K-chunk of 32.  Column n = gate*4 + row, gate order {f,i,u,o}.
// Layout used here (contiguous K, A loaded the same way): lane L holds N=L%16; khalf=L>>4 selects
// K in [khalf*16, khalf*16+16); element e of the v16h is K = khalf*16 + e.
// bpack index: [kc][lane][e]  (kc = K/32 chunk, 32 chunks for K=1024)
// ---------------------------------------------------------------------------
__global__ void qlstm_pack_b(const float* __restrict__ Wf,
                             const float* __restrict__ Wi,
                             const float* __restrict__ Wu,
                             const float* __restrict__ Wo,
                             _Float16* __restrict__ bpack) {
  const int id = blockIdx.x * blockDim.x + threadIdx.x;   // 0..1023
  if (id >= 32 * 32) return;
  const int kc    = id >> 5;
  const int lane  = id & 31;
  const int n     = lane & 15;
  const int khalf = lane >> 4;
  const int gate  = n >> 2;
  const int r     = n & 3;
  const float* W = (gate == 0) ? Wf : (gate == 1) ? Wi : (gate == 2) ? Wu : Wo;
  const float* src = W + r * 1028 + kc * 32 + khalf * 16;
  v16h v;
#pragma unroll
  for (int e = 0; e < 16; ++e) v[e] = (_Float16)src[e];
  ((v16h*)bpack)[kc * 32 + lane] = v;
}

// ---------------------------------------------------------------------------
// Kernel 1: pre[t][n] = (embeds[t] @ Wx(n).T) + b(n), n = gate*4+row, via
// v_wmma_f32_16x16x32_f16.  One wave per 16-row tile of T, 4 waves / block.
// A operand layout (16-bit A 16x32, contiguous K, same order as bpack): lane L holds M = L%16;
// VGPR v, half h -> K = (L>>4)*16 + v*2 + h.
// C/D layout: VGPR r, lane L -> M = r + 8*(L>>4), N = L%16.
// ---------------------------------------------------------------------------
__global__ __launch_bounds__(128) void qlstm_pre_wmma(
    const int*   __restrict__ sentence,
    const float* __restrict__ embedding,
    const _Float16* __restrict__ bpack,
    const float* __restrict__ bf, const float* __restrict__ bi,
    const float* __restrict__ bu, const float* __restrict__ bo,
    float* __restrict__ pre, int T, int ntiles) {
  const int tile = blockIdx.x * 4 + (threadIdx.x >> 5);
  if (tile >= ntiles) return;                 // uniform per wave
  const int lane  = threadIdx.x & 31;
  const int n     = lane & 15;                // A row-in-tile AND B/D column
  const int khalf = lane >> 4;
  const int tbase = tile * 16;

  int row = tbase + n;
  if (row >= T) row = T - 1;                  // per-lane clamp (select, no branch)
  const float* __restrict__ arow = embedding + (long)sentence[row] * 1024;
  const v16h*  __restrict__ bq   = (const v16h*)bpack;

  v8f acc = {};
#pragma unroll 4
  for (int kc = 0; kc < 32; ++kc) {
    v16h a;
#pragma unroll
    for (int v = 0; v < 8; ++v) {
      const int k0 = kc * 32 + (khalf << 4) + (v << 1);
      const float2 f2 = *(const float2*)(arow + k0);
      a[2 * v]     = (_Float16)f2.x;
      a[2 * v + 1] = (_Float16)f2.y;
    }
    const v16h b = bq[kc * 32 + lane];
    acc = __builtin_amdgcn_wmma_f32_16x16x32_f16(
        /*neg_a=*/false, a, /*neg_b=*/false, b,
        /*c_mod=*/(short)0, acc, /*reuse_a=*/false, /*reuse_b=*/false);
  }

  const int gate = n >> 2, r4 = n & 3;
  const float* bp = (gate == 0) ? bf : (gate == 1) ? bi : (gate == 2) ? bu : bo;
  const float bias = bp[r4];

#pragma unroll
  for (int r = 0; r < 8; ++r) {
    const int t = tbase + r + (khalf << 3);
    if (t < T) pre[t * 16 + n] = acc[r] + bias;
  }
}

// ---------------------------------------------------------------------------
// Kernel 2: strictly serial LSTM scan, single lane, everything in registers.
// qlayer closed form (CNOT ring -> XOR of Z bits -> product of cosines):
//   c_w = cos(z_w + th_w)
//   q0 = c1*c2*c3,  q1 = c0*c1,  q2 = c0*c1*c2,  q3 = c0*c1*c2*c3
// ---------------------------------------------------------------------------
__device__ __forceinline__ float fast_sigmoid(float x) {
  return __fdividef(1.0f, 1.0f + __expf(-x));
}
__device__ __forceinline__ float fast_tanh(float x) {
  const float e = __expf(2.0f * x);
  return __fdividef(e - 1.0f, e + 1.0f);
}

__global__ __launch_bounds__(32) void qlstm_scan(
    const float* __restrict__ pre,
    const float* __restrict__ Wf, const float* __restrict__ Wi,
    const float* __restrict__ Wu, const float* __restrict__ Wo,
    const float* __restrict__ th_f, const float* __restrict__ th_i,
    const float* __restrict__ th_u, const float* __restrict__ th_o,
    float* __restrict__ hbuf, int T) {
  if (threadIdx.x != 0) return;

  const float* Wg[4] = {Wf, Wi, Wu, Wo};
  const float* Tg[4] = {th_f, th_i, th_u, th_o};
  float wh[4][4][4], th[4][4];
#pragma unroll
  for (int g = 0; g < 4; ++g) {
#pragma unroll
    for (int j = 0; j < 4; ++j) {
      th[g][j] = Tg[g][j];
#pragma unroll
      for (int k = 0; k < 4; ++k) wh[g][j][k] = Wg[g][j * 1028 + 1024 + k];
    }
  }

  float hx[4] = {0.f, 0.f, 0.f, 0.f};
  float cx[4] = {0.f, 0.f, 0.f, 0.f};

  for (int t = 0; t < T; ++t) {
    {  // pull upcoming pre rows toward the WGP (global_prefetch_b8)
      int tp = t + 8; if (tp >= T) tp = T - 1;
      __builtin_prefetch(pre + tp * 16, 0, 3);
    }
    const float4* pp = (const float4*)(pre + t * 16);
    const float4 p0 = pp[0], p1 = pp[1], p2 = pp[2], p3 = pp[3];
    const float pv[16] = {p0.x, p0.y, p0.z, p0.w, p1.x, p1.y, p1.z, p1.w,
                          p2.x, p2.y, p2.z, p2.w, p3.x, p3.y, p3.z, p3.w};
    float q[4][4];
#pragma unroll
    for (int g = 0; g < 4; ++g) {
      float c[4];
#pragma unroll
      for (int j = 0; j < 4; ++j) {
        float z = pv[g * 4 + j];
        z = fmaf(wh[g][j][0], hx[0], z);
        z = fmaf(wh[g][j][1], hx[1], z);
        z = fmaf(wh[g][j][2], hx[2], z);
        z = fmaf(wh[g][j][3], hx[3], z);
        c[j] = __cosf(z + th[g][j]);
      }
      const float c12 = c[1] * c[2];
      q[g][0] = c12 * c[3];
      q[g][1] = c[0] * c[1];
      q[g][2] = c[0] * c12;
      q[g][3] = q[g][2] * c[3];
    }
#pragma unroll
    for (int j = 0; j < 4; ++j) {
      const float fg = fast_sigmoid(q[0][j]);
      const float ig = fast_sigmoid(q[1][j]);
      const float ug = fast_tanh(q[2][j]);
      const float og = fast_sigmoid(q[3][j]);
      cx[j] = fmaf(fg, cx[j], ig * ug);
      hx[j] = og * fast_tanh(cx[j]);
    }
    ((float4*)hbuf)[t] = make_float4(hx[0], hx[1], hx[2], hx[3]);
  }
}

// ---------------------------------------------------------------------------
// Kernel 3: tag logits + log_softmax.  One wave32 per timestep; lane handles
// tags {lane, lane+32} (TAG=50 -> second tag valid for lane<18).
// ---------------------------------------------------------------------------
__global__ __launch_bounds__(256) void qlstm_tags(
    const float* __restrict__ hbuf, const float* __restrict__ Wt,
    const float* __restrict__ bt, float* __restrict__ out, int T) {
  const int wave = threadIdx.x >> 5;
  const int lane = threadIdx.x & 31;
  const int t = blockIdx.x * 8 + wave;
  if (t >= T) return;  // uniform per wave

  const float4 h = ((const float4*)hbuf)[t];
  const int tag0 = lane, tag1 = lane + 32;
  const bool v1 = (tag1 < 50);

  float l0 = bt[tag0];
  l0 = fmaf(h.x, Wt[tag0 * 4 + 0], l0);
  l0 = fmaf(h.y, Wt[tag0 * 4 + 1], l0);
  l0 = fmaf(h.z, Wt[tag0 * 4 + 2], l0);
  l0 = fmaf(h.w, Wt[tag0 * 4 + 3], l0);

  float l1 = -INFINITY;
  if (v1) {
    l1 = bt[tag1];
    l1 = fmaf(h.x, Wt[tag1 * 4 + 0], l1);
    l1 = fmaf(h.y, Wt[tag1 * 4 + 1], l1);
    l1 = fmaf(h.z, Wt[tag1 * 4 + 2], l1);
    l1 = fmaf(h.w, Wt[tag1 * 4 + 3], l1);
  }

  float m = fmaxf(l0, l1);
#pragma unroll
  for (int off = 16; off >= 1; off >>= 1) m = fmaxf(m, __shfl_xor(m, off, 32));

  float s = __expf(l0 - m) + (v1 ? __expf(l1 - m) : 0.0f);
#pragma unroll
  for (int off = 16; off >= 1; off >>= 1) s += __shfl_xor(s, off, 32);

  const float ls = __logf(s) + m;
  out[t * 50 + tag0] = l0 - ls;
  if (v1) out[t * 50 + tag1] = l1 - ls;
}

// ---------------------------------------------------------------------------
extern "C" void kernel_launch(void* const* d_in, const int* in_sizes, int n_in,
                              void* d_out, int out_size, void* d_ws, size_t ws_size,
                              hipStream_t stream) {
  const int*   sentence  = (const int*)d_in[0];
  const float* embedding = (const float*)d_in[1];
  const float* Wf = (const float*)d_in[2];
  const float* bf = (const float*)d_in[3];
  const float* Wi = (const float*)d_in[4];
  const float* bi = (const float*)d_in[5];
  const float* Wu = (const float*)d_in[6];
  const float* bu = (const float*)d_in[7];
  const float* Wo = (const float*)d_in[8];
  const float* bo = (const float*)d_in[9];
  const float* th_f = (const float*)d_in[10];
  const float* th_i = (const float*)d_in[11];
  const float* th_u = (const float*)d_in[12];
  const float* th_o = (const float*)d_in[13];
  const float* Wt = (const float*)d_in[14];
  const float* bt = (const float*)d_in[15];

  const int T = in_sizes[0];

  char* ws = (char*)d_ws;
  _Float16* bpack = (_Float16*)ws;                               // 32 KB
  float*    pre   = (float*)(ws + 32768);                        // T*16*4 B
  float*    hbuf  = (float*)(ws + 32768 + (size_t)T * 16 * 4);   // T*4*4  B
  float*    out   = (float*)d_out;

  qlstm_pack_b<<<4, 256, 0, stream>>>(Wf, Wi, Wu, Wo, bpack);

  const int ntiles = (T + 15) / 16;
  qlstm_pre_wmma<<<(ntiles + 3) / 4, 128, 0, stream>>>(
      sentence, embedding, bpack, bf, bi, bu, bo, pre, T, ntiles);

  qlstm_scan<<<1, 32, 0, stream>>>(pre, Wf, Wi, Wu, Wo,
                                   th_f, th_i, th_u, th_o, hbuf, T);

  qlstm_tags<<<(T + 7) / 8, 256, 0, stream>>>(hbuf, Wt, bt, out, T);
}
